// GraphUNets_66537633350266
// MI455X (gfx1250) — matrix operand order repaired and checked by reading, not compiled, not run on hardware
//
#include <hip/hip_runtime.h>
#include <hip/hip_bf16.h>

// ---------------------------------------------------------------------------
// Graph U-Net forward on MI455X (gfx1250), wave32 + WMMA f16 (f32 accumulate)
// ---------------------------------------------------------------------------

typedef _Float16 f16;
typedef __attribute__((ext_vector_type(16))) _Float16 v16h;
typedef __attribute__((ext_vector_type(8)))  float    v8f;

#define DEVFN __device__ __forceinline__

DEVFN float eluf(float x) { return x > 0.f ? x : expm1f(x); }

// ---------------------------------------------------------------------------
// WMMA GEMM: C[M,N] = A[M,K] * W[K,N]    (f32 in memory, f16 WMMA, f32 acc)
// CT = column tiles per wave (wave computes a 16 x (CT*16) strip of C).
// grid.x = N/(CT*16) column panels, grid.y covers row tiles, 8 waves/block.
// W panel staged in LDS as f16 *pre-swizzled into WMMA B-fragment order*:
// for each (K-step, col-tile) the 16 elements a lane needs are contiguous
// (32 B, 32 B aligned) -> fragment load is 2x ds_load_b128, not 16x u16.
// A fragment loaded once per K-step and reused across CT WMMAs.
// Requires: M%16==0, K%32==0, N%(CT*16)==0, K<=384.
// ---------------------------------------------------------------------------
template <int CT>
__global__ void __launch_bounds__(256)
gemm_wmma_f16(const float* __restrict__ A, const float* __restrict__ W,
              float* __restrict__ C, int M, int K, int Nout) {
    constexpr int CTW = CT * 16;
    // (K/32 steps) x CT tiles x (32 lanes x 16 f16)
    __shared__ f16 Wlds[(384 / 32) * CT * 512];
    const int col0 = blockIdx.x * CTW;
    // cooperative stage: global (k, c) -> swizzled fragment slot
    for (int idx = threadIdx.x; idx < K * CTW; idx += 256) {
        int k = idx / CTW, c = idx % CTW;
        int step  = k >> 5, kk = k & 31;
        int halfk = (kk >> 3) & 1, w = (kk & 7) + ((kk >> 4) << 3);     // B: e<8 -> k = half*8 + e ; e>=8 -> +16
        int t = c >> 4, r = c & 15;
        Wlds[((step * CT + t) << 9) + (halfk * 16 + r) * 16 + w] =
            (f16)W[k * Nout + col0 + c];
    }
    __syncthreads();

    const int wave = threadIdx.x >> 5;
    const int lane = threadIdx.x & 31;
    const int half = lane >> 4;       // 0: lanes 0-15, 1: lanes 16-31
    const int r    = lane & 15;
    const int tileRow = blockIdx.y * 8 + wave;
    if (tileRow * 16 >= M) return;    // wave-uniform guard (EXEC stays all-1s)

    const float* Arow = A + (size_t)(tileRow * 16 + r) * K;
    v8f acc[CT] = {};
    for (int k0 = 0; k0 < K; k0 += 32) {
        if (k0 + 32 < K) __builtin_prefetch(Arow + k0 + 32, 0, 1);
        v16h a;
#pragma unroll
        for (int w = 0; w < 8; ++w) {
            // A 16x32 f16 layout: e<8 -> k = half*8 + e ; e>=8 -> +16
            a[w]     = (f16)Arow[k0 + half * 8 + w];
            a[w + 8] = (f16)Arow[k0 + 16 + half * 8 + w];
        }
        const int stepBase = ((k0 >> 5) * CT) << 9;
#pragma unroll
        for (int t = 0; t < CT; ++t) {
            v16h b = *(const v16h*)&Wlds[stepBase + (t << 9) + lane * 16];
            acc[t] = __builtin_amdgcn_wmma_f32_16x16x32_f16(
                /*neg_a=*/false, a, /*neg_b=*/false, b,
                /*c_mod=*/(short)0, acc[t], /*reuse_a=*/false, /*reuse_b=*/false);
        }
    }
#pragma unroll
    for (int t = 0; t < CT; ++t) {
#pragma unroll
        for (int e = 0; e < 8; ++e) {
            int orow = tileRow * 16 + e + 8 * half;   // C/D f32 layout
            C[(size_t)orow * Nout + col0 + t * 16 + r] = acc[t][e];
        }
    }
}

// ---------------------------------------------------------------------------
// Elementwise / graph utility kernels
// ---------------------------------------------------------------------------
__global__ void fill_f32_k(float* p, float v, int n) {
    int i = blockIdx.x * blockDim.x + threadIdx.x;
    if (i < n) p[i] = v;
}
__global__ void fill_i32_k(int* p, int v, int n) {
    int i = blockIdx.x * blockDim.x + threadIdx.x;
    if (i < n) p[i] = v;
}
__global__ void elu_k(float* p, int n) {
    int i = blockIdx.x * blockDim.x + threadIdx.x;
    if (i < n) p[i] = eluf(p[i]);
}
__global__ void deg_accum_k(const int* __restrict__ dst, const float* __restrict__ ev,
                            float* deg, int E) {
    int i = blockIdx.x * blockDim.x + threadIdx.x;
    if (i < E) atomicAdd(&deg[dst[i]], ev[i]);
}
__global__ void rsqrt_k(float* d, int n) {
    int i = blockIdx.x * blockDim.x + threadIdx.x;
    if (i < n) d[i] = rsqrtf(d[i]);
}
// out[i,f] = h[i,f] * 2*dinv[i]^2 + b[f]   (GCN self-loop term + bias)
__global__ void selfloop_bias_k(const float* __restrict__ h, const float* __restrict__ dinv,
                                const float* __restrict__ b, float* out, int M, int Fd) {
    int i = blockIdx.x * blockDim.x + threadIdx.x;
    if (i >= M * Fd) return;
    int node = i / Fd, f = i % Fd;
    float di = dinv[node];
    out[i] = h[i] * 2.f * di * di + b[f];
}
// out[dst,f] += h[src,f] * ev * dinv[src]*dinv[dst]
__global__ void edge_agg_k(const int* __restrict__ src, const int* __restrict__ dst,
                           const float* __restrict__ ev, const float* __restrict__ dinv,
                           const float* __restrict__ h, float* out, int E, int Fd) {
    int t = blockIdx.x * blockDim.x + threadIdx.x;
    if (t >= E * Fd) return;
    int e = t / Fd, f = t % Fd;
    float w = ev[e];
    if (w == 0.f) return;
    int s = src[e], d = dst[e];
    atomicAdd(&out[(size_t)d * Fd + f], h[(size_t)s * Fd + f] * w * dinv[s] * dinv[d]);
}
// rn = 1/||p||
__global__ void rnorm_k(const float* __restrict__ p, int len, float* rn) {
    __shared__ float s[128];
    float acc = 0.f;
    for (int i = threadIdx.x; i < len; i += 128) acc += p[i] * p[i];
    s[threadIdx.x] = acc;
    __syncthreads();
    for (int st = 64; st > 0; st >>= 1) {
        if (threadIdx.x < st) s[threadIdx.x] += s[threadIdx.x + st];
        __syncthreads();
    }
    if (threadIdx.x == 0) rn[0] = rsqrtf(s[0]);
}
__global__ void score_k(const float* __restrict__ x, const float* __restrict__ p,
                        const float* __restrict__ rn, float* sc, int M, int Fd) {
    int i = blockIdx.x * blockDim.x + threadIdx.x;
    if (i >= M) return;
    const float* xr = x + (size_t)i * Fd;
    float acc = 0.f;
    for (int f = 0; f < Fd; ++f) acc += xr[f] * p[f];
    sc[i] = tanhf(acc * rn[0]);
}
// per-graph full bitonic sort (descending, tie: lower index first); keep top-k
__global__ void __launch_bounds__(512)
topk_k(const float* __restrict__ score, float* vals, int* perm, int npg, int k) {
    __shared__ float key[1024];
    __shared__ int   kid[1024];
    int g = blockIdx.x;
    for (int i = threadIdx.x; i < npg; i += blockDim.x) {
        key[i] = score[g * npg + i];
        kid[i] = i;
    }
    __syncthreads();
    for (int sz = 2; sz <= npg; sz <<= 1) {
        for (int st = sz >> 1; st > 0; st >>= 1) {
            for (int i = threadIdx.x; i < npg; i += blockDim.x) {
                int j = i ^ st;
                if (j > i) {
                    bool dirDesc = ((i & sz) == 0);
                    float ka = key[i], kb = key[j];
                    int ia = kid[i], ib = kid[j];
                    bool aBefore = (ka > kb) || (ka == kb && ia < ib);
                    bool sw = dirDesc ? !aBefore : aBefore;
                    if (sw) { key[i] = kb; key[j] = ka; kid[i] = ib; kid[j] = ia; }
                }
            }
            __syncthreads();
        }
    }
    for (int j = threadIdx.x; j < k; j += blockDim.x) {
        vals[g * k + j] = key[j];
        perm[g * k + j] = g * npg + kid[j];
    }
}
__global__ void set_remap_k(const int* __restrict__ perm, int* remap, int cnt) {
    int j = blockIdx.x * blockDim.x + threadIdx.x;
    if (j < cnt) remap[perm[j]] = j;
}
__global__ void remap_edges_k(const int* __restrict__ src, const int* __restrict__ dst,
                              const float* __restrict__ ev, const int* __restrict__ remap,
                              int* s2, int* d2, float* ev2, int E) {
    int e = blockIdx.x * blockDim.x + threadIdx.x;
    if (e >= E) return;
    int a = remap[src[e]], b = remap[dst[e]];
    bool valid = (ev[e] != 0.f) && (a >= 0) && (b >= 0);
    s2[e] = a > 0 ? a : 0;
    d2[e] = b > 0 ? b : 0;
    ev2[e] = valid ? 1.f : 0.f;
}
// xn[j,f] = x[perm[j],f] * vals[j]
__global__ void gather_scale_k(const float* __restrict__ x, const int* __restrict__ perm,
                               const float* __restrict__ vals, float* xn, int cnt, int Fd) {
    int t = blockIdx.x * blockDim.x + threadIdx.x;
    if (t >= cnt * Fd) return;
    int j = t / Fd, f = t % Fd;
    xn[t] = x[(size_t)perm[j] * Fd + f] * vals[j];
}
// out[i, 0:leftF) = 0 ; out[i, leftF+c) = right[i,c]
__global__ void concat_fill_k(const float* __restrict__ right, float* out,
                              int M, int leftF, int rightF) {
    int Wd = leftF + rightF;
    int t = blockIdx.x * blockDim.x + threadIdx.x;
    if (t >= M * Wd) return;
    int i = t / Wd, c = t % Wd;
    out[t] = (c < leftF) ? 0.f : right[(size_t)i * rightF + (c - leftF)];
}
// out[perm[j], f] = xs[j, f]  for f < Fs   (left region of width-outF rows)
__global__ void scatter_rows_k(const float* __restrict__ xs, const int* __restrict__ perm,
                               float* out, int cnt, int Fs, int outF) {
    int t = blockIdx.x * blockDim.x + threadIdx.x;
    if (t >= cnt * Fs) return;
    int j = t / Fs, f = t % Fs;
    out[(size_t)perm[j] * outF + f] = xs[t];
}
// per-graph max + mean pool over Npg rows of 32 features, elu, concat -> g0[B,64]
__global__ void __launch_bounds__(256)
pool_k(const float* __restrict__ x, float* g0, int Npg) {
    __shared__ float smax[256], ssum[256];
    int g = blockIdx.x, t = threadIdx.x;
    int f = t & 31, sub = t >> 5;
    float mx = -INFINITY, sm = 0.f;
    for (int r = sub; r < Npg; r += 8) {
        float v = x[((size_t)g * Npg + r) * 32 + f];
        mx = fmaxf(mx, v);
        sm += v;
    }
    smax[t] = mx; ssum[t] = sm;
    __syncthreads();
    for (int st = 128; st >= 32; st >>= 1) {
        if (t < st) { smax[t] = fmaxf(smax[t], smax[t + st]); ssum[t] += ssum[t + st]; }
        __syncthreads();
    }
    if (t < 32) {
        g0[g * 64 + t]      = eluf(smax[t]);
        g0[g * 64 + 32 + t] = eluf(ssum[t] / (float)Npg);
    }
}
// g1 = elu(g0 @ Wl); logits = g1 @ Wc + bc; out = log_softmax(logits)
__global__ void __launch_bounds__(64)
head_k(const float* __restrict__ g0, const float* __restrict__ Wl,
       const float* __restrict__ Wc, const float* __restrict__ bc, float* out) {
    __shared__ float g1[64];
    __shared__ float lg[10];
    __shared__ float red[2];
    int row = blockIdx.x, t = threadIdx.x;
    float acc = 0.f;
    for (int k = 0; k < 64; ++k) acc += g0[row * 64 + k] * Wl[k * 64 + t];
    g1[t] = eluf(acc);
    __syncthreads();
    if (t < 10) {
        float a = bc[t];
        for (int k = 0; k < 64; ++k) a += g1[k] * Wc[k * 10 + t];
        lg[t] = a;
    }
    __syncthreads();
    if (t == 0) {
        float m = lg[0];
        for (int c = 1; c < 10; ++c) m = fmaxf(m, lg[c]);
        float s = 0.f;
        for (int c = 0; c < 10; ++c) s += expf(lg[c] - m);
        red[0] = m; red[1] = logf(s);
    }
    __syncthreads();
    if (t < 10) out[row * 10 + t] = lg[t] - red[0] - red[1];
}

// ---------------------------------------------------------------------------
// Host orchestration
// ---------------------------------------------------------------------------
#define L1D(kern, n, ...) kern<<<((n) + 255) / 256, 256, 0, stream>>>(__VA_ARGS__)

extern "C" void kernel_launch(void* const* d_in, const int* in_sizes, int n_in,
                              void* d_out, int out_size, void* d_ws, size_t ws_size,
                              hipStream_t stream) {
    (void)in_sizes; (void)n_in; (void)out_size; (void)ws_size;

    const int B = 64, N = 1024, F = 128;
    const int E = 64 * 1024 * 8;
    const int K1 = 512, K2 = 256, K3 = 128;
    const int n0 = B * N, n1 = B * K1, n2 = B * K2, n3 = B * K3;

    const float* x  = (const float*)d_in[0];
    const float* W1 = (const float*)d_in[1];  const float* b1 = (const float*)d_in[2];
    const float* p1 = (const float*)d_in[3];
    const float* W2 = (const float*)d_in[4];  const float* b2 = (const float*)d_in[5];
    const float* p2 = (const float*)d_in[6];
    const float* W3 = (const float*)d_in[7];  const float* b3 = (const float*)d_in[8];
    const float* p3 = (const float*)d_in[9];
    const float* W4 = (const float*)d_in[10]; const float* b4 = (const float*)d_in[11];
    const float* W5 = (const float*)d_in[12]; const float* b5 = (const float*)d_in[13];
    const float* W6 = (const float*)d_in[14]; const float* b6 = (const float*)d_in[15];
    const float* W7 = (const float*)d_in[16]; const float* b7 = (const float*)d_in[17];
    const float* Wl = (const float*)d_in[18];
    const float* Wc = (const float*)d_in[19]; const float* bc = (const float*)d_in[20];
    const int* srcE = (const int*)d_in[21];
    const int* dstE = (const int*)d_in[22];

    // bump allocator over workspace
    char* ws = (char*)d_ws;
    size_t off = 0;
    auto alloc = [&](size_t bytes) -> void* {
        void* p = ws + off;
        off = (off + bytes + 255) & ~size_t(255);
        return p;
    };
    float* ev0   = (float*)alloc((size_t)E * 4);
    float* x1    = (float*)alloc((size_t)n0 * 32 * 4);
    float* x3    = (float*)alloc((size_t)n1 * 64 * 4);
    float* x5    = (float*)alloc((size_t)n2 * 128 * 4);
    float* bufA  = (float*)alloc((size_t)n0 * 96 * 4);   // x8 / x10 / x12 (all 6.29M f)
    float* bufB  = (float*)alloc((size_t)2097152 * 4);   // h (all layers exactly 2M f)
    float* bufC  = (float*)alloc((size_t)2097152 * 4);   // x2/x4/x6/x9/x11
    float* bufD  = (float*)alloc((size_t)2097152 * 4);   // x7/x13
    float* deg   = (float*)alloc((size_t)n0 * 4);
    float* score = (float*)alloc((size_t)n0 * 4);
    float* rn    = (float*)alloc(256);
    float* vals2 = (float*)alloc((size_t)n1 * 4);
    float* vals4 = (float*)alloc((size_t)n2 * 4);
    float* vals6 = (float*)alloc((size_t)n3 * 4);
    float* g0    = (float*)alloc((size_t)B * 64 * 4);
    int* perm2 = (int*)alloc((size_t)n1 * 4);
    int* perm4 = (int*)alloc((size_t)n2 * 4);
    int* perm6 = (int*)alloc((size_t)n3 * 4);
    int* remap = (int*)alloc((size_t)n0 * 4);
    int* s2 = (int*)alloc((size_t)E * 4); int* d2 = (int*)alloc((size_t)E * 4);
    int* s4 = (int*)alloc((size_t)E * 4); int* d4 = (int*)alloc((size_t)E * 4);
    int* s6 = (int*)alloc((size_t)E * 4); int* d6 = (int*)alloc((size_t)E * 4);
    float* ev2 = (float*)alloc((size_t)E * 4);
    float* ev4 = (float*)alloc((size_t)E * 4);
    float* ev6 = (float*)alloc((size_t)E * 4);

    auto launch_gemm = [&](const float* A, const float* W, float* C,
                           int M, int K, int Nout) {
        if (Nout % 64 == 0) {
            dim3 grid(Nout / 64, (M / 16 + 7) / 8);
            gemm_wmma_f16<4><<<grid, 256, 0, stream>>>(A, W, C, M, K, Nout);
        } else {
            dim3 grid(Nout / 32, (M / 16 + 7) / 8);
            gemm_wmma_f16<2><<<grid, 256, 0, stream>>>(A, W, C, M, K, Nout);
        }
    };
    auto run_gcn = [&](const float* xin, const float* W, const float* bias,
                       const int* es, const int* ed, const float* eev,
                       int M, int K, int Fout, float* hbuf, float* out) {
        launch_gemm(xin, W, hbuf, M, K, Fout);
        L1D(fill_f32_k, M, deg, 2.0f, M);
        L1D(deg_accum_k, E, ed, eev, deg, E);
        L1D(rsqrt_k, M, deg, M);                              // deg -> dinv
        L1D(selfloop_bias_k, M * Fout, hbuf, deg, bias, out, M, Fout);
        L1D(edge_agg_k, E * Fout, es, ed, eev, deg, hbuf, out, E, Fout);
    };
    auto run_pool = [&](const float* xin, const float* p, int M, int Fd, int npg, int k,
                        const int* esi, const int* edi, const float* eevi,
                        float* vals, int* perm, int* eso, int* edo, float* eevo,
                        float* xout) {
        rnorm_k<<<1, 128, 0, stream>>>(p, Fd, rn);
        L1D(score_k, M, xin, p, rn, score, M, Fd);
        topk_k<<<B, 512, 0, stream>>>(score, vals, perm, npg, k);
        int cnt = B * k;
        L1D(gather_scale_k, cnt * Fd, xin, perm, vals, xout, cnt, Fd);
        L1D(fill_i32_k, M, remap, -1, M);
        L1D(set_remap_k, cnt, perm, remap, cnt);
        L1D(remap_edges_k, E, esi, edi, eevi, remap, eso, edo, eevo, E);
    };

    L1D(fill_f32_k, E, ev0, 1.0f, E);

    // ---- Encoder ----
    run_gcn(x, W1, b1, srcE, dstE, ev0, n0, F, 32, bufB, x1);
    L1D(elu_k, n0 * 32, x1, n0 * 32);                               // x1
    run_pool(x1, p1, n0, 32, N, K1, srcE, dstE, ev0, vals2, perm2, s2, d2, ev2, bufC);
    L1D(elu_k, n1 * 32, bufC, n1 * 32);                             // x2
    run_gcn(bufC, W2, b2, s2, d2, ev2, n1, 32, 64, bufB, x3);
    L1D(elu_k, n1 * 64, x3, n1 * 64);                               // x3
    run_pool(x3, p2, n1, 64, K1, K2, s2, d2, ev2, vals4, perm4, s4, d4, ev4, bufC);
    L1D(elu_k, n2 * 64, bufC, n2 * 64);                             // x4
    run_gcn(bufC, W3, b3, s4, d4, ev4, n2, 64, 128, bufB, x5);
    L1D(elu_k, n2 * 128, x5, n2 * 128);                             // x5
    run_pool(x5, p3, n2, 128, K2, K3, s4, d4, ev4, vals6, perm6, s6, d6, ev6, bufC);
    L1D(elu_k, n3 * 128, bufC, n3 * 128);                           // x6
    run_gcn(bufC, W4, b4, s6, d6, ev6, n3, 128, 256, bufB, bufD);
    L1D(elu_k, n3 * 256, bufD, n3 * 256);                           // x7

    // ---- Decoder ----
    // x8 = elu( concat( scatter(x7 @ perm6), x5 ) )  -> bufA [n2 x 384]
    L1D(concat_fill_k, n2 * 384, x5, bufA, n2, 256, 128);
    L1D(scatter_rows_k, n3 * 256, bufD, perm6, bufA, n3, 256, 384);
    L1D(elu_k, n2 * 384, bufA, n2 * 384);
    run_gcn(bufA, W5, b5, s4, d4, ev4, n2, 384, 128, bufB, bufC);
    L1D(elu_k, n2 * 128, bufC, n2 * 128);                           // x9
    // x10 -> bufA [n1 x 192]
    L1D(concat_fill_k, n1 * 192, x3, bufA, n1, 128, 64);
    L1D(scatter_rows_k, n2 * 128, bufC, perm4, bufA, n2, 128, 192);
    L1D(elu_k, n1 * 192, bufA, n1 * 192);
    run_gcn(bufA, W6, b6, s2, d2, ev2, n1, 192, 64, bufB, bufC);
    L1D(elu_k, n1 * 64, bufC, n1 * 64);                             // x11
    // x12 -> bufA [n0 x 96]
    L1D(concat_fill_k, n0 * 96, x1, bufA, n0, 64, 32);
    L1D(scatter_rows_k, n1 * 64, bufC, perm2, bufA, n1, 64, 96);
    L1D(elu_k, n0 * 96, bufA, n0 * 96);
    run_gcn(bufA, W7, b7, srcE, dstE, ev0, n0, 96, 32, bufB, bufD);
    L1D(elu_k, n0 * 32, bufD, n0 * 32);                             // x13

    // ---- Readout ----
    pool_k<<<B, 256, 0, stream>>>(bufD, g0, N);
    head_k<<<B, 64, 0, stream>>>(g0, Wl, Wc, bc, (float*)d_out);
}
